// SigmaMoE_Acce_57054345560722
// MI455X (gfx1250) — compile-verified, matrix operand order checked by reading, not run on hardware
//
#include <hip/hip_runtime.h>
#include <hip/hip_bf16.h>

// ---------------- problem constants ----------------
#define T_TOK 2048   // B*S
#define HDIM  2048
#define FDIM  1408
#define NEXP  8
#define SFDIM 2816   // shared intermediate
#define MT    16     // token tile (one WMMA M tile)

typedef __attribute__((ext_vector_type(16))) __bf16 v16bf;
typedef __attribute__((ext_vector_type(8)))  float  v8f;
typedef unsigned short u16;
typedef unsigned int   u32;

// fp32 -> bf16 round-to-nearest-even
__device__ __forceinline__ u16 f2bf(float f) {
    u32 u = __float_as_uint(f);
    u32 r = u + 0x7FFFu + ((u >> 16) & 1u);
    return (u16)(r >> 16);
}
__device__ __forceinline__ u32 pk2(float a, float b) {
    return (u32)f2bf(a) | ((u32)f2bf(b) << 16);
}

// Load a 16-element bf16 fragment (two contiguous uint4) from a row-major
// source where this lane's row/col base pointer has already been applied.
__device__ __forceinline__ v16bf ld_frag(const u16* p) {
    union { uint4 q[2]; v16bf v; } f;
    f.q[0] = *(const uint4*)(p);
    f.q[1] = *(const uint4*)(p + 16);
    return f.v;
}

__device__ __forceinline__ v8f wmma_bf16(v16bf a, v16bf b, v8f c) {
    return __builtin_amdgcn_wmma_f32_16x16x32_bf16(
        false, a, false, b, (short)0, c, false, false);
}

__device__ __forceinline__ float silu(float x) {
    return x / (1.0f + __expf(-x));
}

// ---------------- kernel 1: zero routing counters ----------------
__global__ void k_init(int* cnt) {
    if (threadIdx.x < NEXP) cnt[threadIdx.x] = 0;
}

// ---------------- kernel 2: gate = softmax(x @ Wg^T), top-2, build routes ----
__global__ __launch_bounds__(256) void k_gate(
    const float* __restrict__ x, const float* __restrict__ gw,
    int* __restrict__ cnt, int* __restrict__ ridx, float* __restrict__ rwt)
{
    int gtid = blockIdx.x * blockDim.x + threadIdx.x;
    int tok  = gtid >> 5;           // one wave32 per token
    int lane = threadIdx.x & 31;
    if (tok >= T_TOK) return;
    const float* xr = x + (size_t)tok * HDIM;

    float s[NEXP];
    #pragma unroll
    for (int e = 0; e < NEXP; ++e) {
        const float* w = gw + (size_t)e * HDIM;
        float acc = 0.f;
        for (int h = lane; h < HDIM; h += 32) acc = fmaf(xr[h], w[h], acc);
        #pragma unroll
        for (int off = 16; off > 0; off >>= 1) acc += __shfl_xor(acc, off, 32);
        s[e] = acc;
    }
    if (lane == 0) {
        float mx = s[0];
        #pragma unroll
        for (int e = 1; e < NEXP; ++e) mx = fmaxf(mx, s[e]);
        float sum = 0.f;
        #pragma unroll
        for (int e = 0; e < NEXP; ++e) { s[e] = __expf(s[e] - mx); sum += s[e]; }
        float inv = 1.0f / sum;
        int i0 = 0;
        #pragma unroll
        for (int e = 1; e < NEXP; ++e) if (s[e] > s[i0]) i0 = e;
        int i1 = (i0 == 0) ? 1 : 0;
        #pragma unroll
        for (int e = 0; e < NEXP; ++e) if (e != i0 && s[e] > s[i1]) i1 = e;
        int s0 = atomicAdd(&cnt[i0], 1);
        ridx[i0 * T_TOK + s0] = tok;  rwt[i0 * T_TOK + s0] = s[i0] * inv;
        int s1 = atomicAdd(&cnt[i1], 1);
        ridx[i1 * T_TOK + s1] = tok;  rwt[i1 * T_TOK + s1] = s[i1] * inv;
    }
}

// ---------------- kernel 3: transpose + fp32->bf16 ------------------------
// in: [R][C] fp32, out: [C][R] bf16
__global__ __launch_bounds__(256) void k_tc(
    const float* __restrict__ in, u16* __restrict__ out, int R, int C)
{
    __shared__ float tile[32][33];
    int c0 = blockIdx.x * 32, r0 = blockIdx.y * 32;
    int tx = threadIdx.x, ty = threadIdx.y;      // block (32, 8)
    #pragma unroll
    for (int i = 0; i < 32; i += 8) {
        int r = r0 + ty + i, c = c0 + tx;
        tile[ty + i][tx] = (r < R && c < C) ? in[(size_t)r * C + c] : 0.f;
    }
    __syncthreads();
    #pragma unroll
    for (int i = 0; i < 32; i += 8) {
        int c = c0 + ty + i, r = r0 + tx;
        if (c < C && r < R) out[(size_t)c * R + r] = f2bf(tile[tx][ty + i]);
    }
}

// ---------------- shared-expert MLP ---------------------------------------
// y = silu(x@Wg) * (x@Wu) @ Wd ; Wg^T,Wu^T : [SF][H] bf16, Wd^T : [H][SF] bf16
__global__ __launch_bounds__(256) void k_shared(
    const float* __restrict__ x,
    const u16* __restrict__ wgT, const u16* __restrict__ wuT,
    const u16* __restrict__ wdT, float* __restrict__ out)
{
    constexpr int SA = HDIM + 8;    // LDS strides (bf16 elems), keep 16B align
    constexpr int SH = SFDIM + 8;
    extern __shared__ char smem[];
    u16* xs = (u16*)smem;           // [MT][SA]
    u16* hs = xs + MT * SA;         // [MT][SH]

    const int t0   = blockIdx.x * MT;
    const int tid  = threadIdx.x;
    const int wave = tid >> 5;
    const int lane = tid & 31;
    const int mrow = lane & 15;
    const int hsel = (lane >> 4) & 1;

    // stage A tile: 16 rows of x -> bf16 LDS
    for (int i = tid; i < MT * (HDIM / 4); i += 256) {
        int m = i >> 9, h4 = i & 511;
        const float4 v = *(const float4*)(x + (size_t)(t0 + m) * HDIM + 4 * h4);
        uint2 p; p.x = pk2(v.x, v.y); p.y = pk2(v.z, v.w);
        *(uint2*)(xs + m * SA + 4 * h4) = p;
    }
    __syncthreads();

    // phase 1: h[16][SF] = silu(x@Wg) * (x@Wu)
    for (int nc = wave; nc < SFDIM / 16; nc += 8) {
        int n0 = nc * 16;
        v8f ag = {}, au = {};
        const u16* bg = wgT + (size_t)(n0 + mrow) * HDIM + hsel * 8;
        const u16* bu = wuT + (size_t)(n0 + mrow) * HDIM + hsel * 8;
        const u16* ap = xs + mrow * SA + hsel * 8;
        for (int kb = 0; kb < HDIM; kb += 32) {
            v16bf a = ld_frag(ap + kb);
            ag = wmma_bf16(a, ld_frag(bg + kb), ag);
            au = wmma_bf16(a, ld_frag(bu + kb), au);
        }
        #pragma unroll
        for (int r = 0; r < 8; ++r) {
            int M = r + 8 * hsel;
            hs[M * SH + n0 + mrow] = f2bf(silu(ag[r]) * au[r]);
        }
    }
    __syncthreads();

    // phase 2: out[16][H] = h @ Wd
    for (int nc = wave; nc < HDIM / 16; nc += 8) {
        int n0 = nc * 16;
        v8f acc = {};
        const u16* bd = wdT + (size_t)(n0 + mrow) * SFDIM + hsel * 8;
        const u16* ap = hs + mrow * SH + hsel * 8;
        for (int kb = 0; kb < SFDIM; kb += 32)
            acc = wmma_bf16(ld_frag(ap + kb), ld_frag(bd + kb), acc);
        #pragma unroll
        for (int r = 0; r < 8; ++r) {
            int M = r + 8 * hsel;
            out[(size_t)(t0 + M) * HDIM + n0 + mrow] = acc[r];
        }
    }
}

// ---------------- routed expert MLP ----------------------------------------
__global__ __launch_bounds__(256) void k_expert(
    const float* __restrict__ x,
    const u16* __restrict__ wgT,   // [E][F][H]
    const u16* __restrict__ wuT,   // [E][F][H]
    const u16* __restrict__ wdT,   // [E][H][F]
    const int* __restrict__ cnt, const int* __restrict__ ridx,
    const float* __restrict__ rwt, float* __restrict__ out)
{
    constexpr int SA = HDIM + 8;
    constexpr int SH = FDIM + 8;
    const int e    = blockIdx.y;
    const int tile = blockIdx.x;
    const int n    = cnt[e];
    if (tile * MT >= n) return;

    extern __shared__ char smem[];
    u16* xs = (u16*)smem;
    u16* hs = xs + MT * SA;
    __shared__ int   tokr[MT];
    __shared__ float wrow[MT];

    const int tid  = threadIdx.x;
    const int wave = tid >> 5;
    const int lane = tid & 31;
    const int mrow = lane & 15;
    const int hsel = (lane >> 4) & 1;

    if (tid < MT) {
        int r = tile * MT + tid;
        tokr[tid] = (r < n) ? ridx[e * T_TOK + r] : 0;
        wrow[tid] = (r < n) ? rwt[e * T_TOK + r] : 0.0f;
    }
    __syncthreads();

    // gather + convert A tile
    for (int i = tid; i < MT * (HDIM / 4); i += 256) {
        int m = i >> 9, h4 = i & 511;
        const float4 v = *(const float4*)(x + (size_t)tokr[m] * HDIM + 4 * h4);
        uint2 p; p.x = pk2(v.x, v.y); p.y = pk2(v.z, v.w);
        *(uint2*)(xs + m * SA + 4 * h4) = p;
    }
    __syncthreads();

    const size_t wOff = (size_t)e * FDIM * HDIM;

    // phase 1: h = silu(x@Wg_e) * (x@Wu_e)
    for (int nc = wave; nc < FDIM / 16; nc += 8) {
        int n0 = nc * 16;
        v8f ag = {}, au = {};
        const u16* bg = wgT + wOff + (size_t)(n0 + mrow) * HDIM + hsel * 8;
        const u16* bu = wuT + wOff + (size_t)(n0 + mrow) * HDIM + hsel * 8;
        const u16* ap = xs + mrow * SA + hsel * 8;
        for (int kb = 0; kb < HDIM; kb += 32) {
            v16bf a = ld_frag(ap + kb);
            ag = wmma_bf16(a, ld_frag(bg + kb), ag);
            au = wmma_bf16(a, ld_frag(bu + kb), au);
        }
        #pragma unroll
        for (int r = 0; r < 8; ++r) {
            int M = r + 8 * hsel;
            hs[M * SH + n0 + mrow] = f2bf(silu(ag[r]) * au[r]);
        }
    }
    __syncthreads();

    // phase 2: out += w_row * (h @ Wd_e)
    for (int nc = wave; nc < HDIM / 16; nc += 8) {
        int n0 = nc * 16;
        v8f acc = {};
        const u16* bd = wdT + (size_t)e * HDIM * FDIM
                            + (size_t)(n0 + mrow) * FDIM + hsel * 8;
        const u16* ap = hs + mrow * SH + hsel * 8;
        for (int kb = 0; kb < FDIM; kb += 32)
            acc = wmma_bf16(ld_frag(ap + kb), ld_frag(bd + kb), acc);
        #pragma unroll
        for (int r = 0; r < 8; ++r) {
            int M = r + 8 * hsel;
            atomicAdd(out + (size_t)tokr[M] * HDIM + n0 + mrow, acc[r] * wrow[M]);
        }
    }
}

// ---------------- host side -------------------------------------------------
extern "C" void kernel_launch(void* const* d_in, const int* in_sizes, int n_in,
                              void* d_out, int out_size, void* d_ws, size_t ws_size,
                              hipStream_t stream) {
    const float* x   = (const float*)d_in[0];   // [T][H]
    const float* gw  = (const float*)d_in[1];   // [E][H]
    const float* gp  = (const float*)d_in[2];   // [E][H][F]
    const float* up  = (const float*)d_in[3];   // [E][H][F]
    const float* dp  = (const float*)d_in[4];   // [E][F][H]
    const float* sg  = (const float*)d_in[5];   // [H][SF]
    const float* su  = (const float*)d_in[6];   // [H][SF]
    const float* sd  = (const float*)d_in[7];   // [SF][H]
    float* out = (float*)d_out;

    // workspace carve-up (bf16 transposed weights + routing)
    char* p = (char*)d_ws;
    u16* wGT = (u16*)p; p += (size_t)NEXP * FDIM * HDIM * 2;
    u16* wUT = (u16*)p; p += (size_t)NEXP * FDIM * HDIM * 2;
    u16* wDT = (u16*)p; p += (size_t)NEXP * HDIM * FDIM * 2;
    u16* wSG = (u16*)p; p += (size_t)SFDIM * HDIM * 2;
    u16* wSU = (u16*)p; p += (size_t)SFDIM * HDIM * 2;
    u16* wSD = (u16*)p; p += (size_t)HDIM * SFDIM * 2;
    int*   cnt  = (int*)p;   p += 256;
    int*   ridx = (int*)p;   p += (size_t)NEXP * T_TOK * 4;
    float* rwt  = (float*)p; p += (size_t)NEXP * T_TOK * 4;

    k_init<<<1, 32, 0, stream>>>(cnt);
    k_gate<<<(T_TOK * 32) / 256, 256, 0, stream>>>(x, gw, cnt, ridx, rwt);

    dim3 tb(32, 8);
    for (int e = 0; e < NEXP; ++e) {
        k_tc<<<dim3(FDIM / 32, HDIM / 32), tb, 0, stream>>>(
            gp + (size_t)e * HDIM * FDIM, wGT + (size_t)e * FDIM * HDIM, HDIM, FDIM);
        k_tc<<<dim3(FDIM / 32, HDIM / 32), tb, 0, stream>>>(
            up + (size_t)e * HDIM * FDIM, wUT + (size_t)e * FDIM * HDIM, HDIM, FDIM);
        k_tc<<<dim3(HDIM / 32, FDIM / 32), tb, 0, stream>>>(
            dp + (size_t)e * FDIM * HDIM, wDT + (size_t)e * HDIM * FDIM, FDIM, HDIM);
    }
    k_tc<<<dim3(SFDIM / 32, HDIM / 32), tb, 0, stream>>>(sg, wSG, HDIM, SFDIM);
    k_tc<<<dim3(SFDIM / 32, HDIM / 32), tb, 0, stream>>>(su, wSU, HDIM, SFDIM);
    k_tc<<<dim3(HDIM / 32, SFDIM / 32), tb, 0, stream>>>(sd, wSD, SFDIM, HDIM);

    // shared MLP writes every element of d_out (must precede expert atomics)
    size_t shm_shared = (size_t)(MT * (HDIM + 8) + MT * (SFDIM + 8)) * 2;
    k_shared<<<T_TOK / MT, 256, shm_shared, stream>>>(x, wSG, wSU, wSD, out);

    // routed experts accumulate on top
    size_t shm_exp = (size_t)(MT * (HDIM + 8) + MT * (FDIM + 8)) * 2;
    k_expert<<<dim3(T_TOK / MT, NEXP), 256, shm_exp, stream>>>(
        x, wGT, wUT, wDT, cnt, ridx, rwt, out);
}
